// CMP_71236327571847
// MI455X (gfx1250) — matrix operand order repaired and checked by reading, not compiled, not run on hardware
//
#include <hip/hip_runtime.h>
#include <hip/hip_bf16.h>

// ---------------------------------------------------------------------------
// Problem constants (reference: V,C,H,W,E = 2000,16,32,32,4000; C3=48)
// ---------------------------------------------------------------------------
constexpr int kV   = 2000;
constexpr int kC   = 16;
constexpr int kC3  = 48;
constexpr int kHW  = 1024;   // 32*32
constexpr int kE   = 4000;
constexpr int kCP  = 64;     // channels padded to 64 (zero-fill 48..63) for K-chunking

typedef __attribute__((ext_vector_type(16))) __bf16         v16bf;
typedef __attribute__((ext_vector_type(8)))  float          v8f;
typedef __attribute__((ext_vector_type(8)))  unsigned short v8us;
typedef __attribute__((ext_vector_type(16))) unsigned short v16us;

__device__ __forceinline__ unsigned short f2bf(float f) {
  unsigned u = __builtin_bit_cast(unsigned, f);
  unsigned r = u + 0x7FFFu + ((u >> 16) & 1u);
  return (unsigned short)(r >> 16);
}

// ---------------------------------------------------------------------------
// 1) Build x = concat([feats, 0, 0]) channel-wise (pooled slots zeroed)
// ---------------------------------------------------------------------------
__global__ void k_build_x(const float* __restrict__ feats, float* __restrict__ x) {
  size_t idx = (size_t)blockIdx.x * 256 + threadIdx.x;
  size_t total = (size_t)kV * kC3 * kHW;
  if (idx >= total) return;
  int pix = (int)(idx % kHW);
  size_t t = idx / kHW;
  int c = (int)(t % kC3);
  int n = (int)(t / kC3);
  float v = 0.f;
  if (c < kC) v = feats[((size_t)n * kC + c) * kHW + pix];
  x[idx] = v;
}

// ---------------------------------------------------------------------------
// 2) Signed edge scatter: both directions, atomicAdd into pos(16..31)/neg(32..47)
// ---------------------------------------------------------------------------
__global__ void k_scatter(const long long* __restrict__ edges,
                          const float* __restrict__ feats,
                          float* __restrict__ x) {
  int e = blockIdx.x;                 // 0 .. 2E-1 (forward then reversed)
  long long src, sgn, dst;
  if (e < kE) { src = edges[3*e+0]; sgn = edges[3*e+1]; dst = edges[3*e+2]; }
  else { int f = e - kE; src = edges[3*f+2]; sgn = edges[3*f+1]; dst = edges[3*f+0]; }
  int cb = (sgn > 0) ? kC : 2 * kC;   // pooled_pos : pooled_neg
  const float* sp = feats + (size_t)src * kC * kHW;
  float* dp = x + ((size_t)dst * kC3 + cb) * kHW;
  for (int i = threadIdx.x; i < kC * kHW; i += 256) {
    int c = i >> 10, pix = i & 1023;
    atomicAdd(&dp[(size_t)c * kHW + pix], sp[(size_t)c * kHW + pix]);
  }
}

// ---------------------------------------------------------------------------
// 3) Weight conversion: fp32 OIHW -> bf16 [tap][cout][cin_pad64]
// ---------------------------------------------------------------------------
__global__ void k_cvt_w(const float* __restrict__ w, unsigned short* __restrict__ wb,
                        int Cout) {
  int idx = blockIdx.x * 256 + threadIdx.x;
  int total = 9 * Cout * kCP;
  if (idx >= total) return;
  int k = idx % kCP;
  int t = idx / kCP;
  int o = t % Cout;
  int tap = t / Cout;
  float v = (k < kC3) ? w[(size_t)(o * kC3 + k) * 9 + tap] : 0.f;
  wb[idx] = f2bf(v);
}

// ---------------------------------------------------------------------------
// 4) Standalone BN stats (only needed for bn1a over the scattered x)
// ---------------------------------------------------------------------------
__global__ void k_zero(float* p, int n) {
  int i = blockIdx.x * 256 + threadIdx.x;
  if (i < n) p[i] = 0.f;
}

__global__ void k_bnstats(const float* __restrict__ x, float* __restrict__ stats) {
  int n = blockIdx.x / kC3, c = blockIdx.x % kC3;
  const float* p = x + ((size_t)n * kC3 + c) * kHW;
  float s = 0.f, q = 0.f;
  for (int i = threadIdx.x; i < kHW; i += 256) { float v = p[i]; s += v; q += v * v; }
  __shared__ float shs[256], shq[256];
  shs[threadIdx.x] = s; shq[threadIdx.x] = q;
  __syncthreads();
  for (int o = 128; o > 0; o >>= 1) {
    if ((int)threadIdx.x < o) { shs[threadIdx.x] += shs[threadIdx.x + o];
                                shq[threadIdx.x] += shq[threadIdx.x + o]; }
    __syncthreads();
  }
  if (threadIdx.x == 0) {
    atomicAdd(&stats[c],       shs[0]);
    atomicAdd(&stats[kC3 + c], shq[0]);
  }
}

// ---------------------------------------------------------------------------
// 5) Fused BN -> ReLU -> conv3x3 (bf16 WMMA implicit GEMM) + bias + residual
//    + fused output BN-stats (for the NEXT layer) + optional fused instance
//    norm/ReLU (final layer writes d_out directly).
//    One image per block, 8 waves. Dynamic LDS:
//      [0, 131200)           bf16 image [1024][64] + zero row
//      [131200, +64KB)       fp32 output staging (only when do_in=1)
// ---------------------------------------------------------------------------
__global__ __launch_bounds__(256, 1)
void k_conv(const float* __restrict__ in, const float* __restrict__ resid,
            const unsigned short* __restrict__ wbf, const float* __restrict__ bias,
            const float* __restrict__ statsIn, const float* __restrict__ gam,
            const float* __restrict__ bet, float* __restrict__ out,
            float* __restrict__ statsOut, int Cout, int use_bn, int do_in) {
  extern __shared__ unsigned short smem[];
  __shared__ float s_scale[kC3], s_shift[kC3];
  __shared__ float s_sum[kC3], s_sq[kC3];

  // ---- stage 0: fold BN affine; zero channel accumulators ----
  if ((int)threadIdx.x < kC3) {
    float sc = 1.f, sh = 0.f;
    if (use_bn) {
      float cnt  = (float)kV * (float)kHW;
      float mean = statsIn[threadIdx.x] / cnt;
      float var  = statsIn[kC3 + threadIdx.x] / cnt - mean * mean;
      float inv  = rsqrtf(var + 1e-5f);
      sc = gam[threadIdx.x] * inv;
      sh = bet[threadIdx.x] - mean * sc;
    }
    s_scale[threadIdx.x] = sc;
    s_shift[threadIdx.x] = sh;
    s_sum[threadIdx.x] = 0.f;
    s_sq[threadIdx.x]  = 0.f;
  }
  __syncthreads();

  // ---- stage 1: global -> LDS, fused BN+ReLU, fp32 -> bf16 ----
  int img = blockIdx.x;
  const float* xin = in + (size_t)img * kC3 * kHW;
  for (int i = threadIdx.x; i < kCP; i += 256) smem[1024 * kCP + i] = 0; // zero row
  for (int i = threadIdx.x; i < kHW * kCP; i += 256) {
    int c = i >> 10, p = i & 1023;          // coalesced global reads
    unsigned short v = 0;
    if (c < kC3) {
      float f = xin[(size_t)c * kHW + p] * s_scale[c] + s_shift[c];
      if (use_bn) f = fmaxf(f, 0.f);
      v = f2bf(f);
    }
    smem[p * kCP + c] = v;
  }
  __syncthreads();

  // ---- stage 2: 16x16 output tiles via v_wmma_f32_16x16x32_bf16 ----
  int wave = threadIdx.x >> 5;
  int lane = threadIdx.x & 31;
  int half = lane >> 4;                      // wave32 half-select
  int ln   = lane & 15;
  int ntiles = Cout >> 4;
  int tilesTotal = 64 * ntiles;
  float* obuf = (float*)(smem + (kHW + 1) * kCP);   // fp32 staging (do_in)

  for (int tile = wave; tile < tilesTotal; tile += 8) {
    int ntile = tile % ntiles;
    int mtile = tile / ntiles;
    int y0 = mtile >> 1;
    int xA = (mtile & 1) * 16 + ln;          // A-matrix: lane carries M=pixel

    v8f acc = {0.f, 0.f, 0.f, 0.f, 0.f, 0.f, 0.f, 0.f};

#pragma unroll
    for (int tap = 0; tap < 9; ++tap) {
      int dy = tap / 3 - 1, dx = tap % 3 - 1;
      int ys = y0 + dy, xs = xA + dx;
      const unsigned short* arow =
          ((unsigned)ys < 32u && (unsigned)xs < 32u) ? (smem + (ys * 32 + xs) * kCP)
                                                     : (smem + 1024 * kCP);
      const unsigned short* brow =
          wbf + ((size_t)(tap * Cout + ntile * 16 + ln)) * kCP + half * 8;
      __builtin_prefetch(brow, 0, 1);

#pragma unroll
      for (int kc = 0; kc < 2; ++kc) {       // K chunks: channels 0..31, 32..63(pad)
        int cb = kc * 32;
        v8us alo = *(const v8us*)(arow + cb + half * 8);
        v8us ahi = *(const v8us*)(arow + cb + 16 + half * 8);
        v16us au;
#pragma unroll
        for (int j = 0; j < 8; ++j) { au[j] = alo[j]; au[8 + j] = ahi[j]; }
        v8us blo = *(const v8us*)(brow + cb);
        v8us bhi = *(const v8us*)(brow + cb + 16);
        v16us bu;
#pragma unroll
        for (int j = 0; j < 8; ++j) { bu[j] = blo[j]; bu[8 + j] = bhi[j]; }

        v16bf av = __builtin_bit_cast(v16bf, au);
        v16bf bv = __builtin_bit_cast(v16bf, bu);
        acc = __builtin_amdgcn_wmma_f32_16x16x32_bf16(
            false, av, false, bv, (short)0, acc, false, false);
      }
    }

    // ---- epilogue: lane owns channel co x 8 consecutive pixels ----
    int co = ntile * 16 + ln;
    int pbase = mtile * 16 + half * 8;
    float bs = bias[co];
    float vv[8];
#pragma unroll
    for (int r = 0; r < 8; ++r) vv[r] = acc[r] + bs;

    if (resid) {
      const float4* rp = (const float4*)(resid + ((size_t)img * Cout + co) * kHW + pbase);
      float4 ra = rp[0], rb = rp[1];
      vv[0] += ra.x; vv[1] += ra.y; vv[2] += ra.z; vv[3] += ra.w;
      vv[4] += rb.x; vv[5] += rb.y; vv[6] += rb.z; vv[7] += rb.w;
    }
    if (statsOut) {                          // fused BN stats for next layer
      float s = 0.f, q = 0.f;
#pragma unroll
      for (int r = 0; r < 8; ++r) { s += vv[r]; q += vv[r] * vv[r]; }
      atomicAdd(&s_sum[co], s);              // ds_add_f32, low contention
      atomicAdd(&s_sq[co],  q);
    }
    float4 lo = make_float4(vv[0], vv[1], vv[2], vv[3]);
    float4 hi = make_float4(vv[4], vv[5], vv[6], vv[7]);
    if (do_in) {                             // stage in LDS for instance norm
      float4* dp = (float4*)(obuf + co * kHW + pbase);
      dp[0] = lo; dp[1] = hi;
    } else {
      float4* dp = (float4*)(out + ((size_t)img * Cout + co) * kHW + pbase);
      dp[0] = lo; dp[1] = hi;
    }
  }

  // ---- flush fused BN stats once per block ----
  if (statsOut) {
    __syncthreads();
    int t = threadIdx.x;
    if (t < kC3)            atomicAdd(&statsOut[t], s_sum[t]);
    else if (t < 2 * kC3)   atomicAdd(&statsOut[t], s_sq[t - kC3]);
  }

  // ---- fused instance norm + ReLU (final layer): whole image is resident ----
  if (do_in) {
    __syncthreads();
    int c = threadIdx.x >> 4, part = threadIdx.x & 15;  // 16 threads per channel
    float s = 0.f, q = 0.f;
    const float* pc = obuf + c * kHW + part * 64;
    for (int j = 0; j < 64; ++j) { float v = pc[j]; s += v; q += v * v; }
    atomicAdd(&s_sum[c], s);
    atomicAdd(&s_sq[c],  q);
    __syncthreads();
    float* op = out + (size_t)img * kC * kHW;
    for (int i = threadIdx.x; i < kC * kHW; i += 256) {
      int cc = i >> 10;
      float mean = s_sum[cc] * (1.f / kHW);
      float var  = s_sq[cc] * (1.f / kHW) - mean * mean;
      float inv  = rsqrtf(var + 1e-5f);
      op[i] = fmaxf((obuf[i] - mean) * inv, 0.f);   // coalesced store to d_out
    }
  }
}

// ---------------------------------------------------------------------------
// Orchestration
// ---------------------------------------------------------------------------
extern "C" void kernel_launch(void* const* d_in, const int* in_sizes, int n_in,
                              void* d_out, int out_size, void* d_ws, size_t ws_size,
                              hipStream_t stream) {
  const float*     feats  = (const float*)d_in[0];
  const long long* edges  = (const long long*)d_in[1];   // int64 (src,sign,dst)
  const float* bn1a_g = (const float*)d_in[2],  *bn1a_b = (const float*)d_in[3];
  const float* w1a    = (const float*)d_in[4],  *b1a    = (const float*)d_in[5];
  const float* bn1b_g = (const float*)d_in[6],  *bn1b_b = (const float*)d_in[7];
  const float* w1b    = (const float*)d_in[8],  *b1b    = (const float*)d_in[9];
  const float* bn2a_g = (const float*)d_in[10], *bn2a_b = (const float*)d_in[11];
  const float* w2a    = (const float*)d_in[12], *b2a    = (const float*)d_in[13];
  const float* bn2b_g = (const float*)d_in[14], *bn2b_b = (const float*)d_in[15];
  const float* w2b    = (const float*)d_in[16], *b2b    = (const float*)d_in[17];
  const float* wf     = (const float*)d_in[18], *bf     = (const float*)d_in[19];

  const size_t actElems = (size_t)kV * kC3 * kHW;
  float* A = (float*)d_ws;                                // activation ping
  float* B = A + actElems;                                // activation pong
  unsigned short* W = (unsigned short*)(B + actElems);    // bf16 weights
  const size_t wL = (size_t)9 * kC3 * kCP;
  unsigned short* wb1a = W;
  unsigned short* wb1b = wb1a + wL;
  unsigned short* wb2a = wb1b + wL;
  unsigned short* wb2b = wb2a + wL;
  unsigned short* wbf  = wb2b + wL;                       // 9*16*64
  float* statsA = (float*)(wbf + (size_t)9 * kC * kCP);   // 96 floats
  float* statsB = statsA + 2 * kC3;                       // 96 floats

  const int nT = 256;
  const size_t smemConv  = (size_t)(kHW + 1) * kCP * sizeof(unsigned short); // 131,200 B
  const size_t smemFinal = smemConv + (size_t)kC * kHW * sizeof(float);      // +64 KB

  // Build x = [feats | pooled_pos | pooled_neg]
  k_build_x<<<(int)((actElems + 255) / 256), nT, 0, stream>>>(feats, A);
  k_scatter<<<2 * kE, nT, 0, stream>>>(edges, feats, A);

  // bf16 weights
  k_cvt_w<<<(9 * kC3 * kCP + 255) / 256, nT, 0, stream>>>(w1a, wb1a, kC3);
  k_cvt_w<<<(9 * kC3 * kCP + 255) / 256, nT, 0, stream>>>(w1b, wb1b, kC3);
  k_cvt_w<<<(9 * kC3 * kCP + 255) / 256, nT, 0, stream>>>(w2a, wb2a, kC3);
  k_cvt_w<<<(9 * kC3 * kCP + 255) / 256, nT, 0, stream>>>(w2b, wb2b, kC3);
  k_cvt_w<<<(9 * kC  * kCP + 255) / 256, nT, 0, stream>>>(wf,  wbf,  kC);

  // bn1a stats: only standalone stats pass (over scattered x)
  k_zero<<<1, 96, 0, stream>>>(statsA, 96);
  k_bnstats<<<kV * kC3, nT, 0, stream>>>(A, statsA);

  // Resblock 1 (stats for the next BN are produced by each conv's epilogue)
  k_zero<<<1, 96, 0, stream>>>(statsB, 96);
  k_conv<<<kV, nT, smemConv, stream>>>(A, nullptr, wb1a, b1a, statsA,
                                       bn1a_g, bn1a_b, B, statsB, kC3, 1, 0);
  k_zero<<<1, 96, 0, stream>>>(statsA, 96);
  k_conv<<<kV, nT, smemConv, stream>>>(B, A, wb1b, b1b, statsB,
                                       bn1b_g, bn1b_b, A, statsA, kC3, 1, 0);

  // Resblock 2
  k_zero<<<1, 96, 0, stream>>>(statsB, 96);
  k_conv<<<kV, nT, smemConv, stream>>>(A, nullptr, wb2a, b2a, statsA,
                                       bn2a_g, bn2a_b, B, statsB, kC3, 1, 0);
  k_zero<<<1, 96, 0, stream>>>(statsA, 96);
  k_conv<<<kV, nT, smemConv, stream>>>(B, A, wb2b, b2b, statsB,
                                       bn2b_g, bn2b_b, A, statsA, kC3, 1, 0);

  // Final conv 48->16 + fused instance norm + ReLU -> d_out directly
  k_conv<<<kV, nT, smemFinal, stream>>>(A, nullptr, wbf, bf, nullptr,
                                        nullptr, nullptr, (float*)d_out,
                                        nullptr, kC, 0, 1);
}
